// InstoClass_Metric_36713380446637
// MI455X (gfx1250) — matrix operand order repaired and checked by reading, not compiled, not run on hardware
//
#include <hip/hip_runtime.h>

// Problem dims (from reference)
#define B_DIM    64
#define N_CLS    32
#define C_DIM    640
#define L_DIM    441
#define M_DIM    2205
#define R_DIM    (B_DIM * L_DIM)   // 28224 flattened (b,l) rows; 28224 = 64 * 441 exactly
#define RTILES   441               // 28224 / 64 row-tiles of 64
#define MSTEPS   69                // 69*32 = 2208 >= 2205 (32 cols per block step)
#define KSTEPS   20                // 640 / 32
#define LDS_AROW 648               // 640 + 8 f16 pad -> 1296B row stride, b128-aligned

typedef __attribute__((ext_vector_type(16))) _Float16 v16h;
typedef __attribute__((ext_vector_type(8)))  _Float16 v8h;
typedef __attribute__((ext_vector_type(8)))  float    v8f;

union Frag { v16h v; v8h h[2]; };

#define NEGINF (-3.0e38f)

// Branchless sorted-triple insert: 3x v_max_num_f32 + 2x v_min_num_f32, no EXEC churn.
__device__ __forceinline__ void ins3(float v, float& t0, float& t1, float& t2) {
    float d0 = fminf(t0, v);   // value displaced from slot 0
    t0 = fmaxf(t0, v);
    float d1 = fminf(t1, d0);  // value displaced from slot 1
    t1 = fmaxf(t1, d0);
    t2 = fmaxf(t2, d1);
}

// x1 [B, C, l] f32 -> qn [B*l, C] f16, L2-normalized over C per (b,l)
__global__ __launch_bounds__(256) void norm_q_kernel(const float* __restrict__ x1,
                                                     _Float16* __restrict__ qn) {
    int b = blockIdx.y;
    int l = blockIdx.x * 256 + threadIdx.x;
    if (l >= L_DIM) return;
    const float* src = x1 + (size_t)b * C_DIM * L_DIM + l;
    float ss = 0.f;
    for (int c = 0; c < C_DIM; ++c) { float v = src[(size_t)c * L_DIM]; ss += v * v; }
    float r = rsqrtf(ss);
    _Float16* dst = qn + ((size_t)b * L_DIM + l) * C_DIM;
    for (int c = 0; c < C_DIM; ++c) dst[c] = (_Float16)(src[(size_t)c * L_DIM] * r);
}

// x2 [N, C, m] f32 -> sn [N, m, C] f16, L2-normalized over C per (j,m)
__global__ __launch_bounds__(256) void norm_s_kernel(const float* __restrict__ x2,
                                                     _Float16* __restrict__ sn) {
    int j = blockIdx.y;
    int m = blockIdx.x * 256 + threadIdx.x;
    if (m >= M_DIM) return;
    const float* src = x2 + (size_t)j * C_DIM * M_DIM + m;
    float ss = 0.f;
    for (int c = 0; c < C_DIM; ++c) { float v = src[(size_t)c * M_DIM]; ss += v * v; }
    float r = rsqrtf(ss);
    _Float16* dst = sn + ((size_t)j * M_DIM + m) * C_DIM;
    for (int c = 0; c < C_DIM; ++c) dst[c] = (_Float16)(src[(size_t)c * M_DIM] * r);
}

// Fused GEMM + top-3. Grid (441 row-tiles, 32 classes), 256 threads = 8 wave32
// arranged as 4 row-subtiles x 2 col-groups (row-waves share B addresses -> WGP$ hits,
// 4x less L2 traffic than 16-row blocks). rows_out[j*28224 + g] = sum(top3 over m).
__global__ __launch_bounds__(256) void sim_kernel(const _Float16* __restrict__ qn,
                                                  const _Float16* __restrict__ sn,
                                                  float* __restrict__ rows_out) {
    const int rt   = blockIdx.x;       // row tile: 64 flattened (b,l) rows
    const int j    = blockIdx.y;       // class
    const int tid  = threadIdx.x;
    const int w    = tid >> 5;
    const int wm   = w >> 1;           // row-subtile 0..3 (16 rows each)
    const int wn   = w & 1;            // col-group 0..1 (16 cols each)
    const int lane = tid & 31;
    const int half = lane >> 4;
    const int lh   = lane & 15;

    // 83 KB A-tile; 24 KB reduction scratch overlaid after the MACC loop (CDNA5: 320 KB/WGP).
    __shared__ __align__(16) char smem[64 * LDS_AROW * sizeof(_Float16)];
    _Float16* At = (_Float16*)smem;
    float (*red)[32][3] = (float (*)[32][3])smem;   // [64 rows][32 contributors][3]

    // ---- Stage A tile (64 rows x 640 f16) into LDS; 28224 % 64 == 0, no bounds ----
    const int g0 = rt * 64;
    for (int i = tid; i < 64 * 80; i += 256) {      // 80 chunks of 8 f16 per row
        int row = i / 80, ch = i % 80;
        *reinterpret_cast<v8h*>(&At[row * LDS_AROW + ch * 8]) =
            *reinterpret_cast<const v8h*>(qn + ((size_t)(g0 + row)) * C_DIM + ch * 8);
    }
    __syncthreads();

    // ---- Running per-lane top-3 for 8 rows (row = wm*16 + v + 8*half) ----
    float t0[8], t1[8], t2[8];
#pragma unroll
    for (int v = 0; v < 8; ++v) { t0[v] = NEGINF; t1[v] = NEGINF; t2[v] = NEGINF; }

    const int arow = wm * 16 + lh;                  // A row in LDS for this lane

#pragma unroll 1
    for (int mt = 0; mt < MSTEPS; ++mt) {
        const int col  = mt * 32 + wn * 16 + lh;
        const bool cv  = (col < M_DIM);
        const int colc = cv ? col : 0;              // clamp: loads stay in-bounds
        const _Float16* bp = sn + ((size_t)j * M_DIM + colc) * C_DIM;

        // B (32x16 f16): lane%16 = N, contiguous K = half*16 .. +15
        auto loadB = [&](Frag& f, int k) {
            const int kb = k * 32 + half * 16;
            f.h[0] = *reinterpret_cast<const v8h*>(bp + kb);
            f.h[1] = *reinterpret_cast<const v8h*>(bp + kb + 8);
        };

        v8f acc = {0.f, 0.f, 0.f, 0.f, 0.f, 0.f, 0.f, 0.f};
        // 2-deep software pipeline on B so WMMA waits on loadcnt<=4, not 0.
        Frag bq[2];
        loadB(bq[0], 0);
        loadB(bq[1], 1);
#pragma unroll
        for (int k = 0; k < KSTEPS; ++k) {
            Frag bn;
            const bool pf = (k + 2 < KSTEPS);
            if (pf) loadB(bn, k + 2);
            // A (16x32 f16): lane = M, contiguous K = half*16 .. +15
            Frag a;
            const int ka = k * 32 + half * 16;
            a.h[0] = *reinterpret_cast<const v8h*>(&At[arow * LDS_AROW + ka]);
            a.h[1] = *reinterpret_cast<const v8h*>(&At[arow * LDS_AROW + ka + 8]);
            acc = __builtin_amdgcn_wmma_f32_16x16x32_f16(false, a.v, false, bq[k & 1].v,
                                                         (short)0, acc, false, false);
            if (pf) bq[k & 1] = bn;
        }
#pragma unroll
        for (int v = 0; v < 8; ++v) {
            float sc = cv ? acc[v] : NEGINF;        // mask padded columns
            ins3(sc, t0[v], t1[v], t2[v]);
        }
    }

    // ---- Cross-lane / cross-wave top-3 merge via LDS (overlaid on A tile) ----
    __syncthreads();                                // all A reads done before overlay
#pragma unroll
    for (int v = 0; v < 8; ++v) {
        int row = wm * 16 + v + half * 8;           // D layout: M = v + 8*(lane>=16)
        red[row][wn * 16 + lh][0] = t0[v];
        red[row][wn * 16 + lh][1] = t1[v];
        red[row][wn * 16 + lh][2] = t2[v];
    }
    __syncthreads();

    if (tid < 64) {
        float a0 = NEGINF, a1 = NEGINF, a2 = NEGINF;
        for (int i = 0; i < 32; ++i) {
            ins3(red[tid][i][0], a0, a1, a2);
            ins3(red[tid][i][1], a0, a1, a2);
            ins3(red[tid][i][2], a0, a1, a2);
        }
        rows_out[(size_t)j * R_DIM + g0 + tid] = a0 + a1 + a2;
    }
}

// out[b*32+j] = sum over the 441 rows of image b (fixed-order tree => deterministic)
__global__ __launch_bounds__(256) void reduce_kernel(const float* __restrict__ rows_out,
                                                     float* __restrict__ out) {
    const int b = blockIdx.x, j = blockIdx.y, t = threadIdx.x;
    const float* p = rows_out + (size_t)j * R_DIM + (size_t)b * L_DIM;
    float s = (t < L_DIM) ? p[t] : 0.f;
    if (t + 256 < L_DIM) s += p[t + 256];
    __shared__ float sr[256];
    sr[t] = s;
    __syncthreads();
    for (int off = 128; off > 0; off >>= 1) {
        if (t < off) sr[t] += sr[t + off];
        __syncthreads();
    }
    if (t == 0) out[b * N_CLS + j] = sr[0];
}

extern "C" void kernel_launch(void* const* d_in, const int* in_sizes, int n_in,
                              void* d_out, int out_size, void* d_ws, size_t ws_size,
                              hipStream_t stream) {
    const float* x1 = (const float*)d_in[0];   // [64, 640, 441]
    const float* x2 = (const float*)d_in[1];   // [32, 640, 2205]
    float* out = (float*)d_out;                // [64, 32]

    // Workspace layout:
    //   qn       : 28224*640 f16  = 36,126,720 B
    //   sn       : 32*2205*640 f16 = 90,316,800 B
    //   rows_out : 32*28224 f32    =  3,612,672 B   total ~130.1 MB
    char* ws = (char*)d_ws;
    const size_t QN_BYTES = (size_t)R_DIM * C_DIM * sizeof(_Float16);
    const size_t SN_BYTES = (size_t)N_CLS * M_DIM * C_DIM * sizeof(_Float16);
    _Float16* qn       = (_Float16*)ws;
    _Float16* sn       = (_Float16*)(ws + QN_BYTES);
    float*    rows_out = (float*)(ws + QN_BYTES + SN_BYTES);

    norm_q_kernel<<<dim3((L_DIM + 255) / 256, B_DIM), 256, 0, stream>>>(x1, qn);
    norm_s_kernel<<<dim3((M_DIM + 255) / 256, N_CLS), 256, 0, stream>>>(x2, sn);
    sim_kernel<<<dim3(RTILES, N_CLS), 256, 0, stream>>>(qn, sn, rows_out);
    reduce_kernel<<<dim3(B_DIM, N_CLS), 256, 0, stream>>>(rows_out, out);
}
